// ResidualConvAttention_68049461838085
// MI455X (gfx1250) — matrix operand order repaired and checked by reading, not compiled, not run on hardware
//
#include <hip/hip_runtime.h>

typedef __attribute__((ext_vector_type(16))) __bf16    v16bf;
typedef __attribute__((ext_vector_type(8)))  __bf16    v8bf;
typedef __attribute__((ext_vector_type(8)))  float     v8f;
typedef __attribute__((ext_vector_type(4)))  uint32_t  u32x4;
typedef __attribute__((ext_vector_type(8)))  uint32_t  u32x8;

#define USE_TDM 1

// Problem constants (match reference)
constexpr int BATCH    = 4;
constexpr int C_IN     = 256;
constexpr int NPIX     = 128 * 128;     // 16384
constexpr int HEADS    = 8;
constexpr int KDIM     = 64;
constexpr int DK       = HEADS * KDIM;  // 512
constexpr int QKV_ROWS = 3 * DK;        // 1536 (q:0-511, k:512-1023, v:1024-1535)
constexpr float SCALE  = 0.35355339059327373f;  // 64^-0.25
constexpr float LOG2E  = 1.4426950408889634f;
constexpr float EPS    = 1e-5f;

static __device__ __forceinline__ v16bf make_frag(v8bf lo, v8bf hi) {
    v16bf r;
#pragma unroll
    for (int i = 0; i < 8; ++i) { r[i] = lo[i]; r[8 + i] = hi[i]; }
    return r;
}

// ---------------------------------------------------------------------------
// TDM: DMA a 2D bf16 tile (tile_k x tile_rows, row stride in elements) from
// global memory into LDS (rows appended contiguously -> [rows][tile_k]).
// Descriptor layout per CDNA5 ISA 8.3/8.4 (D# group0 = 4 SGPRs, group1 = 8).
// Issued by one wave; completion via TENSORcnt.
// ---------------------------------------------------------------------------
static __device__ __forceinline__ void tdm_load_2d_bf16(
    const __bf16* gaddr, uint32_t lds_byte_addr,
    uint32_t tile_k, uint32_t tile_rows, uint32_t row_stride_elems,
    uint32_t tensor_k, uint32_t tensor_rows) {
    uint64_t ga = (uint64_t)(uintptr_t)gaddr;
    u32x4 g0;
    g0[0] = 1u;                                     // count=1, user descriptor
    g0[1] = lds_byte_addr;                          // lds_addr (bytes)
    g0[2] = (uint32_t)(ga & 0xffffffffu);           // global_addr[31:0]
    g0[3] = (uint32_t)((ga >> 32) & 0x01ffffffu)    // global_addr[56:32]
          | (2u << 30);                             // type=2 ("image")
    u32x8 g1;
    g1[0] = (1u << 16);                             // wg_mask=0, data_size=1 (2B)
    g1[1] = (tensor_k & 0xffffu) << 16;             // ab_addr=0 | tensor_dim0[15:0]
    g1[2] = (tensor_k >> 16) | ((tensor_rows & 0xffffu) << 16); // d0 hi | d1 lo
    g1[3] = (tensor_rows >> 16) | (tile_k << 16);   // d1 hi | tile_dim0
    g1[4] = tile_rows & 0xffffu;                    // tile_dim1 | tile_dim2=0
    g1[5] = row_stride_elems;                       // tensor_dim0_stride[31:0]
    g1[6] = 0u;                                     // d0_stride hi | d1_stride lo
    g1[7] = 0u;                                     // d1_stride hi
    asm volatile("tensor_load_to_lds %0, %1" :: "s"(g0), "s"(g1) : "memory");
}

static __device__ __forceinline__ void wait_tensorcnt0() {
    asm volatile("s_wait_tensorcnt 0x0" ::: "memory");
}

// ---------------------------------------------------------------------------
// Pack weights to bf16: wqkv[1536][256] (q rows then kv rows), wo[256][512]
// ---------------------------------------------------------------------------
__global__ __launch_bounds__(256) void pack_w_kernel(
    const float* __restrict__ wq, const float* __restrict__ wkv,
    const float* __restrict__ wo, __bf16* __restrict__ wqkv_b,
    __bf16* __restrict__ wo_b) {
    int i = blockIdx.x * 256 + threadIdx.x;
    if (i < QKV_ROWS * C_IN) {
        int r = i >> 8, c = i & 255;
        float v = (r < DK) ? wq[i] : wkv[(r - DK) * C_IN + c];
        wqkv_b[i] = (__bf16)v;
    }
    if (i < C_IN * DK) wo_b[i] = (__bf16)wo[i];
}

// x (f32, already [b][c][n] flat) -> bf16
__global__ __launch_bounds__(256) void pack_x_kernel(
    const float* __restrict__ x, __bf16* __restrict__ xb) {
    long i = (long)blockIdx.x * 256 + threadIdx.x;
    xb[i] = (__bf16)x[i];
}

__global__ __launch_bounds__(256) void zero_ctx_kernel(float* __restrict__ ctx) {
    int i = blockIdx.x * 256 + threadIdx.x;
    ctx[i] = 0.0f;
}

// ---------------------------------------------------------------------------
// Generic tiled WMMA GEMM: C[M][N] = A[M][K] x B[K][N] per batch (z).
// A: shared bf16 weights (staged via TDM DMA); B: per-batch bf16 (staged
// transposed in LDS); C: bf16, or f32(+bias).
// BM=128: 8 waves each own one 16-row strip x full 128-col tile.
// ---------------------------------------------------------------------------
template <bool F32OUT>
__global__ __launch_bounds__(256) void gemm_bf16_kernel(
    const __bf16* __restrict__ A, const __bf16* __restrict__ B, long Bbstride,
    void* __restrict__ Cv, long Cbstride, const float* __restrict__ bias,
    int M, int N, int K) {
    constexpr int BM = 128, BN = 128, BK = 32;
    __shared__ __align__(16) __bf16 As[BM * BK];
    __shared__ __align__(16) __bf16 BsT[BN * BK];  // transposed: [n][k]

    const int tid = threadIdx.x, lane = tid & 31, wv = tid >> 5;
    const int n0 = blockIdx.x * BN, m0 = blockIdx.y * BM, bz = blockIdx.z;
    const __bf16* Bb = B + (long)bz * Bbstride;

    v8f acc[8];
#pragma unroll
    for (int t = 0; t < 8; ++t) acc[t] = (v8f){0.f,0.f,0.f,0.f,0.f,0.f,0.f,0.f};

    const int mrow = wv * 16 + (lane & 15);
    const int kb8  = (lane >> 4) * 8;
    const int kb16 = (lane >> 4) * 16;

    for (int k0 = 0; k0 < K; k0 += BK) {
#if USE_TDM
        // Stage A tile (BM x 32) via Tensor Data Mover: one DMA from wave 0,
        // row stride K elements, rows appended contiguously == As[BM][BK].
        if (wv == 0) {
            tdm_load_2d_bf16(A + (long)m0 * K + k0, (uint32_t)(uintptr_t)As,
                             BK, BM, (uint32_t)K, (uint32_t)K, (uint32_t)M);
        }
#else
        // Stage A tile (BM x 32), coalesced b128 loads
#pragma unroll
        for (int l = 0; l < (BM * BK) / (256 * 8); ++l) {
            int chunk = tid + l * 256;
            int ar = chunk >> 2, ac = (chunk & 3) * 8;
            v8bf v = *(const v8bf*)(A + (long)(m0 + ar) * K + k0 + ac);
            *(v8bf*)(As + ar * BK + ac) = v;
        }
#endif
        // Stage B tile (32 x BN), stored transposed in LDS
#pragma unroll
        for (int l = 0; l < (BK * BN) / (256 * 8); ++l) {
            int chunk = tid + l * 256;
            int br = chunk >> 4, bc = (chunk & 15) * 8;
            v8bf v = *(const v8bf*)(Bb + (long)(k0 + br) * N + n0 + bc);
#pragma unroll
            for (int j = 0; j < 8; ++j) BsT[(bc + j) * BK + br] = v[j];
        }
        if (k0 + BK < K) {  // prefetch next B tile rows (global_prefetch_b8)
            __builtin_prefetch(Bb + (long)(k0 + BK + (tid >> 4)) * N + n0 + (tid & 15) * 8, 0, 1);
        }
#if USE_TDM
        if (wv == 0) wait_tensorcnt0();   // DMA landed in LDS (wave 0 tracks it)
#endif
        __syncthreads();                  // publish As (DMA) + BsT to all waves

        // A fragment: 16x32, ISA layout (i<8: K=kb8+i; i>=8: K=16+kb8+i-8)
        v16bf a = make_frag(*(const v8bf*)(As + mrow * BK + kb8),
                            *(const v8bf*)(As + mrow * BK + 16 + kb8));
#pragma unroll
        for (int t = 0; t < 8; ++t) {
            int ncol = t * 16 + (lane & 15);
            // B fragment: 32x16, ISA layout (i<8: K=kb8+i; i>=8: K=16+kb8+i-8) in BsT
            v16bf b = make_frag(*(const v8bf*)(BsT + ncol * BK + kb8),
                                *(const v8bf*)(BsT + ncol * BK + 16 + kb8));
            acc[t] = __builtin_amdgcn_wmma_f32_16x16x32_bf16(
                false, a, false, b, (short)0, acc[t], false, false);
        }
        __syncthreads();  // protect As/BsT before next stage overwrites them
    }

    // Epilogue: D 16x16 f32 layout: row = 8*(lane>>4)+r, col = lane&15
    const int rbase = m0 + wv * 16 + 8 * (lane >> 4);
#pragma unroll
    for (int t = 0; t < 8; ++t) {
        int col = n0 + t * 16 + (lane & 15);
#pragma unroll
        for (int r = 0; r < 8; ++r) {
            int row = rbase + r;
            if constexpr (F32OUT) {
                float* Cf = (float*)Cv + (long)bz * Cbstride;
                Cf[(long)row * N + col] = acc[t][r] + bias[row];
            } else {
                __bf16* Cb = (__bf16*)Cv + (long)bz * Cbstride;
                Cb[(long)row * N + col] = (__bf16)acc[t][r];
            }
        }
    }
}

// ---------------------------------------------------------------------------
// k softmax over n (16384) per (b, row). In place on bf16 qkv rows 512..1023.
// ---------------------------------------------------------------------------
__global__ __launch_bounds__(256) void softmax_k_kernel(__bf16* __restrict__ qkv) {
    __shared__ float red[256];
    const int row = blockIdx.x;                 // b*512 + r
    const int b = row >> 9, r = row & 511;
    __bf16* p = qkv + (long)b * QKV_ROWS * NPIX + (long)(DK + r) * NPIX;
    const int tid = threadIdx.x;

    float m = -1e30f;
    for (int i = tid; i < NPIX; i += 256) m = fmaxf(m, (float)p[i] * SCALE);
    red[tid] = m; __syncthreads();
    for (int s = 128; s > 0; s >>= 1) { if (tid < s) red[tid] = fmaxf(red[tid], red[tid + s]); __syncthreads(); }
    m = red[0]; __syncthreads();

    float sum = 0.f;
    for (int i = tid; i < NPIX; i += 256) sum += exp2f(((float)p[i] * SCALE - m) * LOG2E);
    red[tid] = sum; __syncthreads();
    for (int s = 128; s > 0; s >>= 1) { if (tid < s) red[tid] += red[tid + s]; __syncthreads(); }
    float inv = 1.0f / red[0];

    for (int i = tid; i < NPIX; i += 256)
        p[i] = (__bf16)(exp2f(((float)p[i] * SCALE - m) * LOG2E) * inv);
}

// ---------------------------------------------------------------------------
// q softmax over d (64) per (b,h,n). Coalesced across lanes, 64 regs/thread.
// ---------------------------------------------------------------------------
__global__ __launch_bounds__(256) void softmax_q_kernel(__bf16* __restrict__ qkv) {
    const int col = blockIdx.x * 256 + threadIdx.x;
    const int h = blockIdx.y, b = blockIdx.z;
    __bf16* base = qkv + (long)b * QKV_ROWS * NPIX + (long)(h * KDIM) * NPIX + col;
    float v[64];
    float m = -1e30f;
#pragma unroll
    for (int d = 0; d < 64; ++d) { v[d] = (float)base[(long)d * NPIX] * SCALE; m = fmaxf(m, v[d]); }
    float s = 0.f;
#pragma unroll
    for (int d = 0; d < 64; ++d) { v[d] = exp2f((v[d] - m) * LOG2E); s += v[d]; }
    float inv = 1.0f / s;
#pragma unroll
    for (int d = 0; d < 64; ++d) base[(long)d * NPIX] = (__bf16)(v[d] * inv);
}

// ---------------------------------------------------------------------------
// ctx[b][h][64][64] = k_soft[64 x 16384] x v^T. Split-K (16 slices) + f32
// atomics. Both fragments are contiguous along n -> direct global b128 loads.
// ---------------------------------------------------------------------------
__global__ __launch_bounds__(256) void ctx_kernel(
    const __bf16* __restrict__ qkv, float* __restrict__ ctx) {
    const int bh = blockIdx.y, b = bh >> 3, h = bh & 7;
    const int lane = threadIdx.x & 31, wv = threadIdx.x >> 5;
    const int mt = wv & 3, nt0 = (wv >> 2) * 2;           // 8 waves -> 2 tiles each
    const long base = (long)b * QKV_ROWS * NPIX;
    const __bf16* Krow = qkv + base + (long)(DK + h * KDIM) * NPIX;
    const __bf16* Vrow = qkv + base + (long)(2 * DK + h * KDIM) * NPIX;
    const int kstart = blockIdx.x * (NPIX / 16);          // 1024-wide K slice
    const int d  = mt * 16 + (lane & 15);
    const int e0 = nt0 * 16 + (lane & 15);
    const int kb8 = (lane >> 4) * 8, kb16 = (lane >> 4) * 16;

    v8f c0 = (v8f){0.f,0.f,0.f,0.f,0.f,0.f,0.f,0.f}, c1 = c0;
    for (int k0 = kstart; k0 < kstart + NPIX / 16; k0 += 32) {
        v16bf a = make_frag(*(const v8bf*)(Krow + (long)d * NPIX + k0 + kb8),
                            *(const v8bf*)(Krow + (long)d * NPIX + k0 + 16 + kb8));
        v16bf b0 = make_frag(*(const v8bf*)(Vrow + (long)e0 * NPIX + k0 + kb8),
                             *(const v8bf*)(Vrow + (long)e0 * NPIX + k0 + 16 + kb8));
        v16bf b1 = make_frag(*(const v8bf*)(Vrow + (long)(e0 + 16) * NPIX + k0 + kb8),
                             *(const v8bf*)(Vrow + (long)(e0 + 16) * NPIX + k0 + 16 + kb8));
        c0 = __builtin_amdgcn_wmma_f32_16x16x32_bf16(false, a, false, b0, (short)0, c0, false, false);
        c1 = __builtin_amdgcn_wmma_f32_16x16x32_bf16(false, a, false, b1, (short)0, c1, false, false);
    }
    float* cb = ctx + (long)bh * KDIM * KDIM;
#pragma unroll
    for (int r = 0; r < 8; ++r) {
        int row = mt * 16 + 8 * (lane >> 4) + r;
        atomicAdd(&cb[row * KDIM + nt0 * 16 + (lane & 15)], c0[r]);
        atomicAdd(&cb[row * KDIM + (nt0 + 1) * 16 + (lane & 15)], c1[r]);
    }
}

// ---------------------------------------------------------------------------
// attn[b][h*64+e][n] = sum_d ctx[d][e] * q_soft[d][n]  (A = ctx^T via LDS)
// BM=64 (4 m-strips), 2 n-groups of 64; K=64 in two 32-steps.
// ---------------------------------------------------------------------------
__global__ __launch_bounds__(256) void attn_out_kernel(
    const __bf16* __restrict__ qkv, const float* __restrict__ ctx,
    __bf16* __restrict__ attn) {
    __shared__ __align__(16) __bf16 CT[64 * 64];   // ctx^T: [e][d]
    __shared__ __align__(16) __bf16 BsT[128 * 32]; // q tile transposed [n][k]
    const int tid = threadIdx.x, lane = tid & 31, wv = tid >> 5;
    const int bh = blockIdx.y, b = bh >> 3, h = bh & 7;
    const int n0 = blockIdx.x * 128;
    const int mw = wv & 3, nw = wv >> 2;

    const float* cb = ctx + (long)bh * KDIM * KDIM;
#pragma unroll
    for (int l = 0; l < 16; ++l) {
        int idx = tid + l * 256;
        int dd = idx >> 6, e = idx & 63;
        CT[e * 64 + dd] = (__bf16)cb[idx];
    }
    const __bf16* Qrow = qkv + (long)b * QKV_ROWS * NPIX + (long)(h * KDIM) * NPIX;

    v8f acc[4];
#pragma unroll
    for (int t = 0; t < 4; ++t) acc[t] = (v8f){0.f,0.f,0.f,0.f,0.f,0.f,0.f,0.f};

    const int mrow = mw * 16 + (lane & 15);
    const int kb8 = (lane >> 4) * 8, kb16 = (lane >> 4) * 16;

    for (int k0 = 0; k0 < 64; k0 += 32) {
        __syncthreads();  // protect CT (first iter) / BsT (second iter)
#pragma unroll
        for (int l = 0; l < 2; ++l) {
            int chunk = tid + l * 256;
            int br = chunk >> 4, bc = (chunk & 15) * 8;
            v8bf v = *(const v8bf*)(Qrow + (long)(k0 + br) * NPIX + n0 + bc);
#pragma unroll
            for (int j = 0; j < 8; ++j) BsT[(bc + j) * 32 + br] = v[j];
        }
        __syncthreads();
        v16bf a = make_frag(*(const v8bf*)(CT + mrow * 64 + k0 + kb8),
                            *(const v8bf*)(CT + mrow * 64 + k0 + 16 + kb8));
#pragma unroll
        for (int t = 0; t < 4; ++t) {
            int ncol = nw * 64 + t * 16 + (lane & 15);
            v16bf bf = make_frag(*(const v8bf*)(BsT + ncol * 32 + kb8),
                                 *(const v8bf*)(BsT + ncol * 32 + 16 + kb8));
            acc[t] = __builtin_amdgcn_wmma_f32_16x16x32_bf16(
                false, a, false, bf, (short)0, acc[t], false, false);
        }
    }
    __bf16* ob = attn + (long)b * DK * NPIX + (long)(h * KDIM) * NPIX;
#pragma unroll
    for (int t = 0; t < 4; ++t) {
        int col = n0 + nw * 64 + t * 16 + (lane & 15);
#pragma unroll
        for (int r = 0; r < 8; ++r) {
            int e = mw * 16 + 8 * (lane >> 4) + r;
            ob[(long)e * NPIX + col] = (__bf16)acc[t][r];
        }
    }
}

// ---------------------------------------------------------------------------
// GroupNorm stats per (b, group): groups 0-7 read pre-norm attn-out in d_out,
// groups 8-15 read x directly (concat boundary == group boundary).
// ---------------------------------------------------------------------------
__global__ __launch_bounds__(256) void gn_stats_kernel(
    const float* __restrict__ out_pre, const float* __restrict__ x,
    float* __restrict__ stats) {
    __shared__ float rs[256], rq[256];
    const int g = blockIdx.x, b = blockIdx.y, tid = threadIdx.x;
    const float* base = (g < 8)
        ? out_pre + (long)b * 2 * C_IN * NPIX + (long)(g * 32) * NPIX
        : x       + (long)b * C_IN * NPIX     + (long)((g - 8) * 32) * NPIX;
    const int cnt = 32 * NPIX;
    float s = 0.f, sq = 0.f;
    for (int i = tid; i < cnt; i += 256) { float v = base[i]; s += v; sq += v * v; }
    rs[tid] = s; rq[tid] = sq; __syncthreads();
    for (int w = 128; w > 0; w >>= 1) {
        if (tid < w) { rs[tid] += rs[tid + w]; rq[tid] += rq[tid + w]; }
        __syncthreads();
    }
    if (tid == 0) {
        float mean = rs[0] / (float)cnt;
        float var = rq[0] / (float)cnt - mean * mean;
        stats[(b * 16 + g) * 2 + 0] = mean;
        stats[(b * 16 + g) * 2 + 1] = rsqrtf(var + EPS);
    }
}

__global__ __launch_bounds__(256) void gn_apply_kernel(
    float* __restrict__ out, const float* __restrict__ x,
    const float* __restrict__ stats, const float* __restrict__ gamma,
    const float* __restrict__ beta) {
    long idx = (long)blockIdx.x * 256 + threadIdx.x;   // over b*512*n
    const long per_b = (long)2 * C_IN * NPIX;
    int b = (int)(idx / per_b);
    long rem = idx - (long)b * per_b;
    int ch = (int)(rem / NPIX);
    int col = (int)(rem - (long)ch * NPIX);
    int g = ch >> 5;
    float mean = stats[(b * 16 + g) * 2 + 0];
    float rstd = stats[(b * 16 + g) * 2 + 1];
    float v = (ch < C_IN) ? out[idx]
                          : x[(long)b * C_IN * NPIX + (long)(ch - C_IN) * NPIX + col];
    out[idx] = (v - mean) * rstd * gamma[ch] + beta[ch];
}

// ---------------------------------------------------------------------------
extern "C" void kernel_launch(void* const* d_in, const int* in_sizes, int n_in,
                              void* d_out, int out_size, void* d_ws, size_t ws_size,
                              hipStream_t stream) {
    const float* x     = (const float*)d_in[0];
    const float* wq    = (const float*)d_in[1];
    const float* wkv   = (const float*)d_in[2];
    const float* wo    = (const float*)d_in[3];
    const float* bo    = (const float*)d_in[4];
    const float* gamma = (const float*)d_in[5];
    const float* beta  = (const float*)d_in[6];
    float* out = (float*)d_out;

    // Workspace layout (bytes, all 4KB aligned)
    char* ws = (char*)d_ws;
    __bf16* wqkv_b = (__bf16*)(ws + 0);                      // 1536*256*2  = 768 KB
    __bf16* wo_b   = (__bf16*)(ws + 786432);                 // 256*512*2   = 256 KB
    __bf16* xb     = (__bf16*)(ws + 1048576);                // 4*256*16384*2  = 32 MB
    __bf16* qkv_b  = (__bf16*)(ws + 34603008);               // 4*1536*16384*2 = 192 MB
    float*  ctx    = (float*)(ws + 235929600);               // 4*8*64*64*4 = 512 KB
    __bf16* attn_b = (__bf16*)(ws + 236453888);              // 4*512*16384*2  = 64 MB
    float*  stats  = (float*)(ws + 303562752);               // 4*16*2*4

    // 1. Pack weights + activations to bf16; zero ctx accumulator
    pack_w_kernel<<<(QKV_ROWS * C_IN) / 256, 256, 0, stream>>>(wq, wkv, wo, wqkv_b, wo_b);
    pack_x_kernel<<<(BATCH * C_IN * NPIX) / 256, 256, 0, stream>>>(x, xb);
    zero_ctx_kernel<<<(BATCH * HEADS * KDIM * KDIM) / 256, 256, 0, stream>>>(ctx);

    // 2. qkv projection: [1536 x 16384] = Wqkv[1536x256] x x[256x16384] per batch
    gemm_bf16_kernel<false><<<dim3(NPIX / 128, QKV_ROWS / 128, BATCH), 256, 0, stream>>>(
        wqkv_b, xb, (long)C_IN * NPIX, qkv_b, (long)QKV_ROWS * NPIX, nullptr,
        QKV_ROWS, NPIX, C_IN);

    // 3. Softmaxes (in place, bf16)
    softmax_k_kernel<<<BATCH * DK, 256, 0, stream>>>(qkv_b);
    softmax_q_kernel<<<dim3(NPIX / 256, HEADS, BATCH), 256, 0, stream>>>(qkv_b);

    // 4. ctx = k_soft x v^T per head (split-K=16 + f32 atomics)
    ctx_kernel<<<dim3(16, BATCH * HEADS), 256, 0, stream>>>(qkv_b, ctx);

    // 5. attn = ctx^T x q_soft per head
    attn_out_kernel<<<dim3(NPIX / 128, BATCH * HEADS), 256, 0, stream>>>(qkv_b, ctx, attn_b);

    // 6. out2 = wo x attn + bo -> f32 into d_out channels [0,256) (pre-norm)
    gemm_bf16_kernel<true><<<dim3(NPIX / 128, C_IN / 128, BATCH), 256, 0, stream>>>(
        wo_b, attn_b, (long)DK * NPIX, out, (long)2 * C_IN * NPIX, bo,
        C_IN, NPIX, DK);

    // 7. GroupNorm over concat(out2, x): stats then apply
    gn_stats_kernel<<<dim3(16, BATCH), 256, 0, stream>>>(out, x, stats);
    gn_apply_kernel<<<(int)(((long)BATCH * 2 * C_IN * NPIX) / 256), 256, 0, stream>>>(
        out, x, stats, gamma, beta);
}
